// MemModule_9637906612316
// MI455X (gfx1250) — compile-verified, matrix operand order checked by reading, not run on hardware
//
#include <hip/hip_runtime.h>
#include <stdint.h>

// ---------------- problem constants ----------------
constexpr int N_ = 32, C_ = 256, H_ = 32, W_ = 32, M_ = 2000;
constexpr int MP = 2016;          // M padded to multiple of 32 (K of GEMM2)
constexpr int TT = 32;            // tokens per block (= one (n,h) row, w=0..31)
constexpr int XPITCH = 264;       // bf16 elems per xa row (16B-aligned rows)
constexpr int APITCH = 2020;      // f32 elems per att row (16B-aligned rows)
constexpr int YPITCH = 260;       // f32 elems per staged-y row
constexpr float LAMBD = 0.0025f;
constexpr float EPS   = 1e-12f;

typedef __attribute__((ext_vector_type(16))) __bf16 v16bf;
typedef __attribute__((ext_vector_type(8)))  float  v8f;

union FragU { uint32_t u[8]; v16bf v; };

__device__ __forceinline__ uint16_t f2bf(float f) {   // RNE f32 -> bf16 bits
  uint32_t u = __float_as_uint(f);
  u += 0x7FFFu + ((u >> 16) & 1u);
  return (uint16_t)(u >> 16);
}
__device__ __forceinline__ uint32_t pk2(float a, float b) {
  return (uint32_t)f2bf(a) | ((uint32_t)f2bf(b) << 16);
}
__device__ __forceinline__ v16bf mkfrag(uint4 lo, uint4 hi) {
  FragU f;
  f.u[0] = lo.x; f.u[1] = lo.y; f.u[2] = lo.z; f.u[3] = lo.w;
  f.u[4] = hi.x; f.u[5] = hi.y; f.u[6] = hi.z; f.u[7] = hi.w;
  return f.v;
}
__device__ __forceinline__ v16bf cvtfrag(const float* p) { // 16 f32 -> bf16 frag
  float4 f0 = *(const float4*)(p);
  float4 f1 = *(const float4*)(p + 4);
  float4 f2 = *(const float4*)(p + 16);
  float4 f3 = *(const float4*)(p + 20);
  FragU f;
  f.u[0] = pk2(f0.x, f0.y); f.u[1] = pk2(f0.z, f0.w);
  f.u[2] = pk2(f1.x, f1.y); f.u[3] = pk2(f1.z, f1.w);
  f.u[4] = pk2(f2.x, f2.y); f.u[5] = pk2(f2.z, f2.w);
  f.u[6] = pk2(f3.x, f3.y); f.u[7] = pk2(f3.z, f3.w);
  return f.v;
}
#define WMMA_BF16(A, B, Cacc) \
  __builtin_amdgcn_wmma_f32_16x16x32_bf16(false, (A), false, (B), (short)0, (Cacc), false, false)

// ---------------- kernel 0: weight f32 -> bf16, two layouts ----------------
// wb : [M][C]  row-major  (B operand of GEMM1, K = C contiguous)
// wt : [C][MP] transposed (B operand of GEMM2, K = M contiguous, zero-padded)
__global__ void prep_weights(const float* __restrict__ w,
                             uint16_t* __restrict__ wb,
                             uint16_t* __restrict__ wt) {
  int idx = blockIdx.x * 256 + threadIdx.x;
  if (idx < M_ * C_) {
    int m = idx / C_, c = idx % C_;
    uint16_t b = f2bf(w[idx]);
    wb[idx] = b;
    wt[(size_t)c * MP + m] = b;
  }
  if (idx < C_ * (MP - M_)) {            // zero K-padding rows of wt
    int c = idx / (MP - M_), mp = M_ + idx % (MP - M_);
    wt[(size_t)c * MP + mp] = 0;
  }
}

// ---------------- fused kernel: GEMM1 -> softmax/shrink/L1 -> GEMM2 --------
__launch_bounds__(256)
__global__ void memae_fused(const float* __restrict__ x,
                            const uint16_t* __restrict__ wb,
                            const uint16_t* __restrict__ wt,
                            float* __restrict__ y_out,
                            float* __restrict__ att_out) {
  extern __shared__ char smem[];
  uint16_t* xa  = (uint16_t*)smem;                         // [TT][XPITCH] bf16
  float*    att = (float*)(smem + TT * XPITCH * 2);        // [TT][APITCH] f32
  float*    rsum = att + TT * APITCH;                      // [TT] 1/L1 per token

  const int tid   = threadIdx.x;
  const int lane  = tid & 31;
  const int wv    = tid >> 5;          // 8 waves
  const int hlane = lane & 15;
  const int kb    = (lane < 16) ? 0 : 8;   // ISA 16-bit A/B lane K-base

  const int blk = blockIdx.x;          // over N*H = 1024
  const int n   = blk >> 5;
  const int h   = blk & 31;
  const float* xbase = x + (size_t)n * C_ * H_ * W_ + h * W_; // + c*1024 + w

  // ---- phase 1: load x tile (coalesced, read-once -> NT), bf16 to LDS -----
  for (int c = wv; c < C_; c += 8) {
    float v = __builtin_nontemporal_load(xbase + (size_t)c * H_ * W_ + lane);
    xa[lane * XPITCH + c] = f2bf(v);   // lane == w == token
  }
  // zero att K-padding columns (read by GEMM2)
  for (int i = tid; i < TT * (MP - M_); i += 256)
    att[(i / (MP - M_)) * APITCH + M_ + (i % (MP - M_))] = 0.f;
  __syncthreads();

  // ---- phase 2: GEMM1  logits(32 x 2000) = Xbf16 @ Wbf16^T ----------------
  // 125 column tiles of 16; each wave owns tiles mt = wv, wv+8, ...
  for (int mt = wv; mt < 125; mt += 8) {
    const int m0 = mt * 16;
    v8f acc0 = {}, acc1 = {};
    const uint16_t* brow = wb + (size_t)(m0 + hlane) * C_;  // lane = column
    if (m0 + 128 + hlane < M_)                              // stream next tile
      __builtin_prefetch(wb + (size_t)(m0 + 128 + hlane) * C_, 0, 1);
#pragma unroll
    for (int kc = 0; kc < 8; ++kc) {
      const int k0 = kc * 32 + kb;
      v16bf bf = mkfrag(*(const uint4*)(brow + k0),
                        *(const uint4*)(brow + k0 + 16));
      const uint16_t* a0p = xa + hlane * XPITCH + k0;        // tokens 0..15
      const uint16_t* a1p = xa + (16 + hlane) * XPITCH + k0; // tokens 16..31
      v16bf a0 = mkfrag(*(const uint4*)a0p, *(const uint4*)(a0p + 16));
      v16bf a1 = mkfrag(*(const uint4*)a1p, *(const uint4*)(a1p + 16));
      acc0 = WMMA_BF16(a0, bf, acc0);
      acc1 = WMMA_BF16(a1, bf, acc1);
    }
#pragma unroll
    for (int v = 0; v < 8; ++v) {                     // D: lane&15=col, v=row
      int r = (lane < 16) ? v : v + 8;
      att[r * APITCH + m0 + hlane]        = acc0[v];
      att[(16 + r) * APITCH + m0 + hlane] = acc1[v];
    }
  }
  __syncthreads();

  // ---- phase 3: per-token softmax + hard-shrink-relu, one wave per row ----
  for (int r = wv; r < TT; r += 8) {
    float* row = att + r * APITCH;
    float mx = -3.0e38f;
    for (int m = lane; m < M_; m += 32) mx = fmaxf(mx, row[m]);
    for (int off = 16; off; off >>= 1) mx = fmaxf(mx, __shfl_xor(mx, off, 32));
    float s1 = 0.f;
    for (int m = lane; m < M_; m += 32) {
      float e = __expf(row[m] - mx);
      row[m] = e; s1 += e;
    }
    for (int off = 16; off; off >>= 1) s1 += __shfl_xor(s1, off, 32);
    float inv1 = __frcp_rn(s1);
    float s2 = 0.f;
    for (int m = lane; m < M_; m += 32) {
      float a = row[m] * inv1;
      float d = a - LAMBD;
      float o = fmaxf(d, 0.f) * a / (fabsf(d) + EPS);
      row[m] = o; s2 += o;
    }
    for (int off = 16; off; off >>= 1) s2 += __shfl_xor(s2, off, 32);
    if (lane == 0) rsum[r] = 1.0f / fmaxf(s2, EPS);
  }
  __syncthreads();

  // ---- phase 4: L1-scale in LDS + coalesced NT att output (NMHW) ----------
  // att output is 262 MB write-once: non-temporal keeps weights resident in L2
  float* attg = att_out + (size_t)n * M_ * H_ * W_ + h * W_;
  for (int m = wv; m < M_; m += 8) {
    float v = att[lane * APITCH + m] * rsum[lane];   // lane == token == w
    att[lane * APITCH + m] = v;
    __builtin_nontemporal_store(v, attg + (size_t)m * H_ * W_ + lane);
  }
  __syncthreads();

  // ---- phase 5: GEMM2  y(32 x 256) = att_bf16 @ W_bf16 (K = 2016) ---------
  // Single K loop, 4 live accumulators (2 column tiles x 2 row groups):
  // A fragments converted once per kc and reused by both column tiles.
  const int c0lo = wv * 16, c0hi = (wv + 8) * 16;    // 16 col tiles / 8 waves
  v8f acc00 = {}, acc01 = {}, acc10 = {}, acc11 = {};
  {
    const uint16_t* bt0 = wt + (size_t)(c0lo + hlane) * MP;
    const uint16_t* bt1 = wt + (size_t)(c0hi + hlane) * MP;
    const float* a0row = att + hlane * APITCH;
    const float* a1row = att + (16 + hlane) * APITCH;
    for (int kc = 0; kc < 63; ++kc) {
      const int k0 = kc * 32 + kb;
      v16bf a0 = cvtfrag(a0row + k0);                 // tokens 0..15
      v16bf a1 = cvtfrag(a1row + k0);                 // tokens 16..31
      v16bf b0 = mkfrag(*(const uint4*)(bt0 + k0),
                        *(const uint4*)(bt0 + k0 + 16));
      v16bf b1 = mkfrag(*(const uint4*)(bt1 + k0),
                        *(const uint4*)(bt1 + k0 + 16));
      acc00 = WMMA_BF16(a0, b0, acc00);
      acc01 = WMMA_BF16(a1, b0, acc01);
      acc10 = WMMA_BF16(a0, b1, acc10);
      acc11 = WMMA_BF16(a1, b1, acc11);
    }
  }
  __syncthreads();                 // everyone done reading att -> reuse as y

  float* ys = att;                 // [TT][YPITCH] staging, fits in att region
#pragma unroll
  for (int v = 0; v < 8; ++v) {
    int r = (lane < 16) ? v : v + 8;
    ys[r * YPITCH + c0lo + hlane]        = acc00[v];
    ys[(16 + r) * YPITCH + c0lo + hlane] = acc01[v];
    ys[r * YPITCH + c0hi + hlane]        = acc10[v];
    ys[(16 + r) * YPITCH + c0hi + hlane] = acc11[v];
  }
  __syncthreads();

  // coalesced NT store of y (NCHW, w contiguous per channel; write-once)
  float* yg = y_out + (size_t)n * C_ * H_ * W_ + h * W_;
  for (int c = wv; c < C_; c += 8)
    __builtin_nontemporal_store(ys[lane * YPITCH + c],
                                yg + (size_t)c * H_ * W_ + lane);
}

// ---------------- launch ----------------
extern "C" void kernel_launch(void* const* d_in, const int* in_sizes, int n_in,
                              void* d_out, int out_size, void* d_ws, size_t ws_size,
                              hipStream_t stream) {
  const float* x = (const float*)d_in[0];
  const float* w = (const float*)d_in[1];
  float* y   = (float*)d_out;
  float* att = y + (size_t)N_ * C_ * H_ * W_;     // outputs concatenated (y, att)

  uint16_t* wb = (uint16_t*)d_ws;                 // [M][C] bf16
  uint16_t* wt = wb + (size_t)M_ * C_;            // [C][MP] bf16

  prep_weights<<<(M_ * C_ + 255) / 256, 256, 0, stream>>>(w, wb, wt);

  size_t smem = (size_t)TT * XPITCH * 2           // xa
              + (size_t)TT * APITCH * 4           // att / y staging
              + (size_t)TT * 4;                   // rsum
  memae_fused<<<N_ * H_, 256, smem, stream>>>(x, wb, wt, y, att);
}
